// Quantizer_12094627905954
// MI455X (gfx1250) — compile-verified, matrix operand order checked by reading, not run on hardware
//
#include <hip/hip_runtime.h>
#include <hip/hip_bf16.h>

typedef __attribute__((ext_vector_type(16))) _Float16 v16h;
typedef __attribute__((ext_vector_type(8)))  _Float16 v8h;
typedef __attribute__((ext_vector_type(8)))  float    v8f;

#define BSZ   8
#define NSEQ  4096
#define DD    64
#define KC    8192
#define MTOT  (BSZ * NSEQ)      // 32768 rows of z
#define CHUNK 128               // codes staged in LDS per step
#define NCHUNK (KC / CHUNK)     // 64
#define WAVES 8
#define TPB   256

#if __has_builtin(__builtin_amdgcn_global_load_async_to_lds_b128)
#define HAVE_ASYNC_LDS 1
#else
#define HAVE_ASYNC_LDS 0
#endif

#if HAVE_ASYNC_LDS
typedef __attribute__((ext_vector_type(4))) int i4;
typedef __attribute__((address_space(1))) i4* g_i4p;   // global-AS int4*
typedef __attribute__((address_space(3))) i4* l_i4p;   // LDS-AS int4*
#endif

// -------- combine two contiguous 8-half runs into a WMMA 16-half fragment ----
__device__ __forceinline__ v16h mk16(v8h lo, v8h hi) {
    v16h a;
#pragma unroll
    for (int i = 0; i < 8; ++i) { a[i] = lo[i]; a[8 + i] = hi[i]; }
    return a;
}

// -------- prep: L2-normalize rows of a (rows x 64) f32 matrix to f16 ---------
__global__ void normalize_rows_f16(const float* __restrict__ src,
                                   _Float16* __restrict__ dst,
                                   _Float16* __restrict__ dstT,
                                   int rows, int ldT) {
    int r = blockIdx.x * blockDim.x + threadIdx.x;
    if (r >= rows) return;
    const float* p = src + (size_t)r * DD;
    float s = 0.f;
#pragma unroll
    for (int i = 0; i < DD; ++i) { float v = p[i]; s += v * v; }
    float n   = fmaxf(sqrtf(s), 1e-12f);
    float inv = 1.0f / n;
#pragma unroll
    for (int i = 0; i < DD; ++i) {
        _Float16 h = (_Float16)(p[i] * inv);
        dst[(size_t)r * DD + i] = h;
        if (dstT) dstT[(size_t)i * ldT + r] = h;
    }
}

// -------- stage one 128-code chunk of the codebook into LDS (both layouts) ---
__device__ __forceinline__ void stage_chunk(int chunk, int tid,
                                            const _Float16* __restrict__ cb,
                                            const _Float16* __restrict__ cbT,
                                            _Float16* lds_cb,
                                            _Float16* lds_cbT) {
    const uint4* src = (const uint4*)(cb + (size_t)chunk * CHUNK * DD);
    uint4*       dst = (uint4*)lds_cb;
    const int row = tid >> 2, seg = tid & 3;
    const uint4* srcT = (const uint4*)(cbT + (size_t)row * KC +
                                       (size_t)chunk * CHUNK + seg * 32);
    uint4* dstT = (uint4*)(lds_cbT + row * CHUNK + seg * 32);
#if HAVE_ASYNC_LDS
    // Direct memory->LDS DMA path (ASYNCcnt), no VGPR round-trip.
#pragma unroll
    for (int k = 0; k < 4; ++k)
        __builtin_amdgcn_global_load_async_to_lds_b128(
            (g_i4p)(src + tid + k * TPB), (l_i4p)(dst + tid + k * TPB), 0, 0);
#pragma unroll
    for (int k = 0; k < 4; ++k)
        __builtin_amdgcn_global_load_async_to_lds_b128(
            (g_i4p)(srcT + k), (l_i4p)(dstT + k), 0, 0);
#if __has_builtin(__builtin_amdgcn_s_wait_asynccnt)
    __builtin_amdgcn_s_wait_asynccnt(0);
#else
    asm volatile("s_wait_asynccnt 0" ::: "memory");
#endif
#else
    // Fallback: load to VGPRs, store to LDS.
#pragma unroll
    for (int k = 0; k < 4; ++k) dst[tid + k * TPB] = src[tid + k * TPB];
#pragma unroll
    for (int k = 0; k < 4; ++k) dstT[k] = srcT[k];
#endif
}

// -------- fused: sim -> softmax (shift = -invT, exact) -> probs + z_q --------
// logits are (sim - 1) * invT with sim = cosine similarity in [-1, 1], so the
// exponent is always <= 0: no running max / rescale / divergent branch needed.
__global__ __launch_bounds__(TPB) void quantizer_main(
    const _Float16* __restrict__ zf,     // (MTOT x 64) normalized z, f16
    const _Float16* __restrict__ cb,     // (KC x 64) normalized codebook, f16
    const _Float16* __restrict__ cbT,    // (64 x KC) transposed codebook, f16
    const float* __restrict__ tempPtr,   // temperature scalar
    float* __restrict__ probs,           // (MTOT x KC)
    float* __restrict__ zq)              // (MTOT x 64)
{
    __shared__ alignas(16) _Float16 lds_cb [CHUNK * DD];     // 16 KB
    __shared__ alignas(16) _Float16 lds_cbT[DD * CHUNK];     // 16 KB
    __shared__ alignas(16) _Float16 lds_ex [WAVES][16 * 32]; // 8 KB, P exchange

    const int tid  = threadIdx.x;
    const int wave = tid >> 5;
    const int lane = tid & 31;
    const int lm   = lane & 15;   // N index inside tile / A-row index
    const int h    = lane >> 4;   // half-wave select

    const float invT = 1.0f / tempPtr[0];
    const float negT = -invT;
    const int m0 = (blockIdx.x * WAVES + wave) * 16;   // this wave's row tile

    // ---- A fragments for this wave's 16 z-rows (kept in VGPRs all kernel) ---
    const v8h* zr = (const v8h*)(zf + (size_t)(m0 + lm) * DD);
    const v16h a0 = mk16(zr[h],     zr[2 + h]);   // K = 0..31
    const v16h a1 = mk16(zr[4 + h], zr[6 + h]);   // K = 32..63

    // ======================= PASS A: sum of exp (branch-free) ================
    float s8[8];
#pragma unroll
    for (int r = 0; r < 8; ++r) s8[r] = 0.f;

    for (int chunk = 0; chunk < NCHUNK; ++chunk) {
        __syncthreads();
        stage_chunk(chunk, tid, cb, cbT, lds_cb, lds_cbT);
        __syncthreads();
        if (chunk + 1 < NCHUNK)
            __builtin_prefetch(cb + (size_t)(chunk + 1) * CHUNK * DD + tid * 16, 0, 1);

        for (int t = 0; t < CHUNK / 16; ++t) {
            const v8h* rp = (const v8h*)(lds_cb + (t * 16 + lm) * DD);
            v16h b0 = mk16(rp[h],     rp[2 + h]);
            v16h b1 = mk16(rp[4 + h], rp[6 + h]);
            v8f c;
#pragma unroll
            for (int i = 0; i < 8; ++i) c[i] = 0.0f;
            c = __builtin_amdgcn_wmma_f32_16x16x32_f16(false, a0, false, b0,
                                                       (short)0, c, false, false);
            c = __builtin_amdgcn_wmma_f32_16x16x32_f16(false, a1, false, b1,
                                                       (short)0, c, false, false);
#pragma unroll
            for (int r = 0; r < 8; ++r)
                s8[r] += __expf(fmaf(c[r], invT, negT));   // exp((sim-1)/T) <= 1
        }
    }
    // sum across the 16 lanes that hold the softmax (N) axis
#pragma unroll
    for (int r = 0; r < 8; ++r) {
        float s = s8[r];
#pragma unroll
        for (int off = 8; off; off >>= 1) s += __shfl_xor(s, off, 32);
        s8[r] = 1.0f / s;   // reciprocal partition function for row r + 8h
    }

    // ============== PASS B: recompute sim, emit probs, build z_q =============
    v8f acc[4];
#pragma unroll
    for (int g = 0; g < 4; ++g)
#pragma unroll
        for (int i = 0; i < 8; ++i) acc[g][i] = 0.0f;

    _Float16* ex = &lds_ex[wave][0];

    for (int chunk = 0; chunk < NCHUNK; ++chunk) {
        __syncthreads();
        stage_chunk(chunk, tid, cb, cbT, lds_cb, lds_cbT);
        __syncthreads();
        if (chunk + 1 < NCHUNK)
            __builtin_prefetch(cbT + (size_t)(tid >> 2) * KC +
                               (size_t)(chunk + 1) * CHUNK + (tid & 3) * 32, 0, 1);

        for (int cc = 0; cc < 4; ++cc) {          // 32-code sub-chunks
            // -- recompute two 16-wide sim tiles, write probs, stash P in LDS -
#pragma unroll
            for (int tt = 0; tt < 2; ++tt) {
                const int t = cc * 2 + tt;
                const v8h* rp = (const v8h*)(lds_cb + (t * 16 + lm) * DD);
                v16h b0 = mk16(rp[h],     rp[2 + h]);
                v16h b1 = mk16(rp[4 + h], rp[6 + h]);
                v8f c;
#pragma unroll
                for (int i = 0; i < 8; ++i) c[i] = 0.0f;
                c = __builtin_amdgcn_wmma_f32_16x16x32_f16(false, a0, false, b0,
                                                           (short)0, c, false, false);
                c = __builtin_amdgcn_wmma_f32_16x16x32_f16(false, a1, false, b1,
                                                           (short)0, c, false, false);
                const size_t ng = (size_t)chunk * CHUNK + t * 16 + lm;
#pragma unroll
                for (int r = 0; r < 8; ++r) {
                    float p = __expf(fmaf(c[r], invT, negT)) * s8[r];
                    probs[(size_t)(m0 + r + 8 * h) * KC + ng] = p;
                    ex[(r + 8 * h) * 32 + tt * 16 + lm] = (_Float16)p;
                }
            }
            // -- reload P in A-fragment layout, accumulate z_q += P @ CB ------
            const v8h* pr = (const v8h*)(ex + lm * 32);
            v16h ap = mk16(pr[h], pr[2 + h]);
#pragma unroll
            for (int g = 0; g < 4; ++g) {
                const v8h* tp = (const v8h*)(lds_cbT + (16 * g + lm) * CHUNK);
                v16h bg = mk16(tp[cc * 4 + h], tp[cc * 4 + 2 + h]);
                acc[g] = __builtin_amdgcn_wmma_f32_16x16x32_f16(false, ap, false, bg,
                                                                (short)0, acc[g],
                                                                false, false);
            }
        }
    }

    // ---- write z_q tile (16 x 64), coalesced 64-B runs ----------------------
#pragma unroll
    for (int g = 0; g < 4; ++g)
#pragma unroll
        for (int r = 0; r < 8; ++r)
            zq[(size_t)(m0 + r + 8 * h) * DD + 16 * g + lm] = acc[g][r];
}

extern "C" void kernel_launch(void* const* d_in, const int* in_sizes, int n_in,
                              void* d_out, int out_size, void* d_ws, size_t ws_size,
                              hipStream_t stream) {
    (void)in_sizes; (void)n_in; (void)out_size; (void)ws_size;
    const float* z    = (const float*)d_in[0];   // (8, 4096, 64)
    const float* emb  = (const float*)d_in[1];   // (8192, 64)
    const float* temp = (const float*)d_in[2];   // scalar

    float* out   = (float*)d_out;
    float* probs = out;                          // (32768 x 8192)
    float* zq    = out + (size_t)MTOT * KC;      // (32768 x 64)

    _Float16* cb_f16  = (_Float16*)d_ws;                 // 1 MiB
    _Float16* cbT_f16 = cb_f16  + (size_t)KC * DD;       // 1 MiB
    _Float16* z_f16   = cbT_f16 + (size_t)KC * DD;       // 4 MiB

    normalize_rows_f16<<<KC / 256,   256, 0, stream>>>(emb, cb_f16, cbT_f16, KC, KC);
    normalize_rows_f16<<<MTOT / 256, 256, 0, stream>>>(z,   z_f16,  nullptr, MTOT, 0);
    quantizer_main<<<MTOT / (WAVES * 16), TPB, 0, stream>>>(z_f16, cb_f16, cbT_f16,
                                                            temp, probs, zq);
}
